// Net_38843684225866
// MI455X (gfx1250) — compile-verified, matrix operand order checked by reading, not run on hardware
//
#include <hip/hip_runtime.h>
#include <math.h>

// ---------------------------------------------------------------------------
// RandLA-Net-ish point cloud classifier for MI455X (gfx1250, wave32).
// Dense layers: LDS-staged tiles -> v_wmma_f32_16x16x32_f16, 16x(16*NT)
// output tile per wave (NT=1 for Co<=16 layers, NT=2 otherwise; A reused by
// NT WMMAs). KNN: LDS position tiles + register insertion network.
// BN: deterministic 2-stage sliced batch-stat reduction.
// ---------------------------------------------------------------------------

#define KNN_K     16
#define KNN_TS    128
#define BN_EPS    1e-6f
#define L_SLOPE   0.2f
#define BN_SLICES 32

typedef __attribute__((ext_vector_type(16))) _Float16 v16h;
typedef __attribute__((ext_vector_type(8)))  float    v8f;

// ---------------------------------------------------------------------------
// GEMM: Y[M,Co] = X[M,Ci] @ W[Ci,Co] (+ bias). REQUIRES M % 16 == 0 (true for
// every call site here). One wave computes a 16 x (NT*16) output tile: A tile
// staged in LDS once per K-chunk and reused by NT WMMAs. K-padding handled by
// zeroing LDS for partial chunks; no per-element predication.
// ---------------------------------------------------------------------------
template <int NT>
__global__ __launch_bounds__(32) void gemm_wmma_kernel(
    const float* __restrict__ X, const float* __restrict__ W,
    const float* __restrict__ bias, float* __restrict__ Y,
    int M, int Ci, int Co)
{
    constexpr int NW = NT * 16;
    __shared__ float As[16][33];        // 16 rows x 32 k (+pad)
    __shared__ float Bs[32][NW + 1];    // 32 k x NW n (+pad)

    const int lane = threadIdx.x;
    const int l15  = lane & 15;
    const int grp  = lane >> 4;
    const int m0   = blockIdx.x * 16;
    const int n0   = blockIdx.y * NW;

    v8f acc[NT];
#pragma unroll
    for (int t = 0; t < NT; ++t) acc[t] = (v8f){0.f,0.f,0.f,0.f,0.f,0.f,0.f,0.f};

    for (int kk = 0; kk < Ci; kk += 32) {
        const int cw = (Ci - kk) < 32 ? (Ci - kk) : 32;

        if (cw < 32) {
            // partial K chunk: zero-pad the tiles first
            for (int t = lane; t < 16 * 32; t += 32) As[t >> 5][t & 31] = 0.f;
            for (int t = lane; t < 32 * NW; t += 32) Bs[t / NW][t % NW] = 0.f;
        }
        __syncthreads();

        if (cw == 32) {
            // full chunk fast path: unconditional loads
            for (int t = lane; t < 16 * 32; t += 32) {
                int r = t >> 5, k = t & 31;
                As[r][k] = X[(size_t)(m0 + r) * Ci + kk + k];
            }
            for (int t = lane; t < 32 * NW; t += 32) {
                int k = t / NW, n = t % NW;
                int c = n0 + n;
                Bs[k][n] = (c < Co) ? W[(size_t)(kk + k) * Co + c] : 0.f;
            }
        } else {
            for (int t = lane; t < 16 * cw; t += 32) {
                int r = t / cw, k = t % cw;
                As[r][k] = X[(size_t)(m0 + r) * Ci + kk + k];
            }
            for (int t = lane; t < cw * NW; t += 32) {
                int k = t / NW, n = t % NW;
                int c = n0 + n;
                if (c < Co) Bs[k][n] = W[(size_t)(kk + k) * Co + c];
            }
        }
        __syncthreads();

        if (kk + 32 < Ci) {
            // speculative prefetch of next A chunk (global_prefetch_b8)
            __builtin_prefetch(&X[(size_t)(m0 + l15) * Ci + kk + 32], 0, 1);
        }

        v16h a;
#pragma unroll
        for (int i = 0; i < 16; ++i) {
            // half-index i -> K per ISA 16-bit A 16x32 layout:
            // K = (i%8) + 8*(2*(i/8) + lane/16)
            int k = (i & 7) + 8 * (((i >> 3) << 1) + grp);
            a[i] = (_Float16)As[l15][k];
        }
#pragma unroll
        for (int t = 0; t < NT; ++t) {
            v16h b;
#pragma unroll
            for (int i = 0; i < 16; ++i) {
                int k = (i & 7) + 8 * (((i >> 3) << 1) + grp);
                b[i] = (_Float16)Bs[k][t * 16 + l15];
            }
            acc[t] = __builtin_amdgcn_wmma_f32_16x16x32_f16(
                false, a, false, b, (short)0, acc[t], false, false);
        }
        __syncthreads();
    }

#pragma unroll
    for (int t = 0; t < NT; ++t) {
        int c = n0 + t * 16 + l15;
        if (c < Co) {
            float bb = bias ? bias[c] : 0.f;
#pragma unroll
            for (int r = 0; r < 8; ++r) {
                int mr = m0 + r + 8 * grp;         // D: M = vgpr + 8*(lane/16)
                Y[(size_t)mr * Co + c] = acc[t][r] + bb;
            }
        }
    }
}

// ---------------------------------------------------------------------------
// BatchNorm batch statistics over dim 0 (biased variance). Deterministic
// 2-stage reduction: (C x BN_SLICES) partial blocks, then per-channel merge.
// ---------------------------------------------------------------------------
__global__ __launch_bounds__(256) void bn_partial_kernel(
    const float* __restrict__ Y, float* __restrict__ ps,
    float* __restrict__ ps2, int M, int C)
{
    const int c = blockIdx.x;
    const int s = blockIdx.y;
    const int t = threadIdx.x;
    const int chunk = (M + BN_SLICES - 1) / BN_SLICES;
    int r0 = s * chunk;
    int r1 = r0 + chunk; if (r1 > M) r1 = M;

    float a = 0.f, a2 = 0.f;
    for (int i = r0 + t; i < r1; i += 256) {
        float v = Y[(size_t)i * C + c];
        a += v; a2 += v * v;
    }
    __shared__ float sh[256], sh2[256];
    sh[t] = a; sh2[t] = a2;
    __syncthreads();
    for (int off = 128; off > 0; off >>= 1) {
        if (t < off) { sh[t] += sh[t + off]; sh2[t] += sh2[t + off]; }
        __syncthreads();
    }
    if (t == 0) {
        ps[c * BN_SLICES + s]  = sh[0];
        ps2[c * BN_SLICES + s] = sh2[0];
    }
}

__global__ void bn_finalize_kernel(const float* __restrict__ ps,
    const float* __restrict__ ps2, float* __restrict__ mean,
    float* __restrict__ rsig, int M, int C)
{
    int c = blockIdx.x * blockDim.x + threadIdx.x;
    if (c >= C) return;
    float s = 0.f, s2 = 0.f;
    for (int i = 0; i < BN_SLICES; ++i) {
        s  += ps[c * BN_SLICES + i];
        s2 += ps2[c * BN_SLICES + i];
    }
    float m = s / (float)M;
    float v = s2 / (float)M - m * m;
    mean[c] = m;
    rsig[c] = rsqrtf(v + BN_EPS);
}

__global__ void bn_apply_kernel(float* __restrict__ Y,
    const float* __restrict__ mean, const float* __restrict__ rsig,
    const float* __restrict__ g, const float* __restrict__ beta,
    int M, int C, int act)
{
    size_t i = (size_t)blockIdx.x * blockDim.x + threadIdx.x;
    if (i >= (size_t)M * C) return;
    int c = (int)(i % C);
    float y = (Y[i] - mean[c]) * rsig[c] * g[c] + beta[c];
    if (act) y = y > 0.f ? y : L_SLOPE * y;
    Y[i] = y;
}

// ---------------------------------------------------------------------------
// Per-cloud self-inclusive KNN (k=16 smallest sq-distances). LDS position
// tiles; register insertion network (fully unrolled -> stays in VGPRs).
// Requires n % KNN_TS == 0 (2048, 512 both OK).
// ---------------------------------------------------------------------------
__global__ __launch_bounds__(KNN_TS) void knn_kernel(
    const float* __restrict__ pos, int* __restrict__ nbr, int n)
{
    const int gid   = blockIdx.x * KNN_TS + threadIdx.x;
    const int cloud = gid / n;
    const int base  = cloud * n;

    const float px = pos[3 * gid + 0];
    const float py = pos[3 * gid + 1];
    const float pz = pos[3 * gid + 2];

    float bd[KNN_K];
    int   bi[KNN_K];
#pragma unroll
    for (int q = 0; q < KNN_K; ++q) { bd[q] = 3.0e38f; bi[q] = gid; }

    __shared__ float sx[KNN_TS], sy[KNN_TS], sz[KNN_TS];

    for (int j0 = 0; j0 < n; j0 += KNN_TS) {
        int js = base + j0 + threadIdx.x;
        sx[threadIdx.x] = pos[3 * js + 0];
        sy[threadIdx.x] = pos[3 * js + 1];
        sz[threadIdx.x] = pos[3 * js + 2];
        __syncthreads();
        for (int j = 0; j < KNN_TS; ++j) {
            float dx = sx[j] - px, dy = sy[j] - py, dz = sz[j] - pz;
            float d = dx * dx + dy * dy + dz * dz;
            if (d < bd[KNN_K - 1]) {
                float dd = d; int jj = base + j0 + j;
#pragma unroll
                for (int p = 0; p < KNN_K; ++p) {
                    if (dd < bd[p]) {
                        float td = bd[p]; int ti = bi[p];
                        bd[p] = dd; bi[p] = jj;
                        dd = td; jj = ti;
                    }
                }
            }
        }
        __syncthreads();
    }
#pragma unroll
    for (int q = 0; q < KNN_K; ++q) nbr[(size_t)gid * KNN_K + q] = bi[q];
}

// ---------------------------------------------------------------------------
// Relative point-position encoding: rel[e] = [pi, pj, pj-pi, |pj-pi|]  (10)
// ---------------------------------------------------------------------------
__global__ void rel_kernel(const float* __restrict__ pos,
    const int* __restrict__ nbr, float* __restrict__ rel, int E)
{
    int e = blockIdx.x * blockDim.x + threadIdx.x;
    if (e >= E) return;
    int i = e / KNN_K;
    int j = nbr[e];
    float pix = pos[3 * i + 0], piy = pos[3 * i + 1], piz = pos[3 * i + 2];
    float pjx = pos[3 * j + 0], pjy = pos[3 * j + 1], pjz = pos[3 * j + 2];
    float dx = pjx - pix, dy = pjy - piy, dz = pjz - piz;
    float dist = sqrtf(dx * dx + dy * dy + dz * dz);
    float* r = rel + (size_t)e * 10;
    r[0] = pix; r[1] = piy; r[2] = piz;
    r[3] = pjx; r[4] = pjy; r[5] = pjz;
    r[6] = dx;  r[7] = dy;  r[8] = dz;  r[9] = dist;
}

// local[e, 0:dh] = x[nbr[e]]; local[e, dh:2dh] = locse[e]
__global__ void concat_kernel(const float* __restrict__ x,
    const float* __restrict__ locse, const int* __restrict__ nbr,
    float* __restrict__ local, int E, int dh)
{
    size_t idx = (size_t)blockIdx.x * blockDim.x + threadIdx.x;
    if (idx >= (size_t)E * dh) return;
    int e = (int)(idx / dh);
    int c = (int)(idx % dh);
    int j = nbr[e];
    float* L = local + (size_t)e * 2 * dh;
    L[c]      = x[(size_t)j * dh + c];
    L[dh + c] = locse[(size_t)e * dh + c];
}

// Attentive pooling: out[m,c] = sum_k softmax_k(att[m,k,c]) * local[m,k,c]
__global__ void attpool_kernel(const float* __restrict__ local,
    const float* __restrict__ att, float* __restrict__ out, int M, int d)
{
    size_t idx = (size_t)blockIdx.x * blockDim.x + threadIdx.x;
    if (idx >= (size_t)M * d) return;
    int m = (int)(idx / d);
    int c = (int)(idx % d);
    const float* A = att   + (size_t)m * KNN_K * d + c;
    const float* L = local + (size_t)m * KNN_K * d + c;
    float mx = -3.0e38f;
#pragma unroll
    for (int k = 0; k < KNN_K; ++k) mx = fmaxf(mx, A[(size_t)k * d]);
    float se = 0.f, acc = 0.f;
#pragma unroll
    for (int k = 0; k < KNN_K; ++k) {
        float w = __expf(A[(size_t)k * d] - mx);
        se  += w;
        acc += w * L[(size_t)k * d];
    }
    out[idx] = acc / se;
}

__global__ void resid_leaky_kernel(const float* __restrict__ a,
    const float* __restrict__ b, float* __restrict__ o, size_t N)
{
    size_t i = (size_t)blockIdx.x * blockDim.x + threadIdx.x;
    if (i >= N) return;
    float v = a[i] + b[i];
    o[i] = v > 0.f ? v : L_SLOPE * v;
}

// Strided decimation gather: out row r -> src row (r/nd)*n + (r%nd)*4
__global__ void decim_kernel(const float* __restrict__ src,
    float* __restrict__ dst, int n, int nd, int C, int Mout)
{
    size_t i = (size_t)blockIdx.x * blockDim.x + threadIdx.x;
    if (i >= (size_t)Mout * C) return;
    int r = (int)(i / C);
    int c = (int)(i % C);
    int srow = (r / nd) * n + (r % nd) * 4;
    dst[i] = src[(size_t)srow * C + c];
}

__global__ void maxpool_kernel(const float* __restrict__ in,
    float* __restrict__ out, int nrow, int Bc, int C)
{
    int i = blockIdx.x * blockDim.x + threadIdx.x;
    if (i >= Bc * C) return;
    int b = i / C, c = i % C;
    float mx = -3.0e38f;
    for (int r = 0; r < nrow; ++r)
        mx = fmaxf(mx, in[((size_t)b * nrow + r) * C + c]);
    out[i] = mx;
}

__global__ void logsoftmax_kernel(const float* __restrict__ in,
    float* __restrict__ out, int R, int C)
{
    int r = blockIdx.x * blockDim.x + threadIdx.x;
    if (r >= R) return;
    const float* v = in + (size_t)r * C;
    float mx = v[0];
    for (int c = 1; c < C; ++c) mx = fmaxf(mx, v[c]);
    float s = 0.f;
    for (int c = 0; c < C; ++c) s += __expf(v[c] - mx);
    float l = __logf(s);
    for (int c = 0; c < C; ++c) out[(size_t)r * C + c] = v[c] - mx - l;
}

// ---------------------------------------------------------------------------
// Host-side orchestration
// ---------------------------------------------------------------------------
struct SmlpP { const float *W, *b, *g, *beta; };
struct LfaP  { SmlpP enc; const float* attW; SmlpP post; };
struct BlockP { SmlpP mlp1, shortcut; LfaP lfa1, lfa2; SmlpP mlp2; };

struct Scratch {
    int* nbr;
    float *rel, *locse, *local, *att, *pooled;
    float *mean, *rsig, *psum, *psum2;
};

static inline unsigned cdiv(size_t a, size_t b) { return (unsigned)((a + b - 1) / b); }

static void run_linear(const float* X, const float* W, const float* b, float* Y,
                       int M, int Ci, int Co, hipStream_t s)
{
    if (Co <= 16) {
        dim3 g(cdiv(M, 16), cdiv(Co, 16));   // 16x16 tile: no dead columns
        gemm_wmma_kernel<1><<<g, dim3(32), 0, s>>>(X, W, b, Y, M, Ci, Co);
    } else {
        dim3 g(cdiv(M, 16), cdiv(Co, 32));   // 16x32 tile: A reused by 2 WMMAs
        gemm_wmma_kernel<2><<<g, dim3(32), 0, s>>>(X, W, b, Y, M, Ci, Co);
    }
}

static void run_smlp(const float* X, const SmlpP& p, float* Y, int M, int Ci,
                     int Co, bool act, const Scratch& w, hipStream_t s)
{
    run_linear(X, p.W, p.b, Y, M, Ci, Co, s);
    bn_partial_kernel<<<dim3((unsigned)Co, BN_SLICES), dim3(256), 0, s>>>(
        Y, w.psum, w.psum2, M, Co);
    bn_finalize_kernel<<<dim3(1), dim3(128), 0, s>>>(
        w.psum, w.psum2, w.mean, w.rsig, M, Co);
    size_t tot = (size_t)M * Co;
    bn_apply_kernel<<<dim3(cdiv(tot, 256)), dim3(256), 0, s>>>(
        Y, w.mean, w.rsig, p.g, p.beta, M, Co, act ? 1 : 0);
}

static void run_lfa(const LfaP& p, const int* nbr, const float* x,
                    const float* pos, float* out, int M, int d,
                    const Scratch& w, hipStream_t s)
{
    const int dh = d / 2;
    const int E = M * KNN_K;
    rel_kernel<<<dim3(cdiv(E, 256)), dim3(256), 0, s>>>(pos, nbr, w.rel, E);
    run_smlp(w.rel, p.enc, w.locse, E, 10, dh, true, w, s);
    size_t tot = (size_t)E * dh;
    concat_kernel<<<dim3(cdiv(tot, 256)), dim3(256), 0, s>>>(
        x, w.locse, nbr, w.local, E, dh);
    run_linear(w.local, p.attW, nullptr, w.att, E, d, d, s);
    size_t md = (size_t)M * d;
    attpool_kernel<<<dim3(cdiv(md, 256)), dim3(256), 0, s>>>(
        w.local, w.att, w.pooled, M, d);
    run_smlp(w.pooled, p.post, out, M, d, d, true, w, s);
}

static void run_block(const BlockP& bp, const float* x, const float* pos,
                      float* x_dec, float* pos_dec, int M, int n,
                      int di, int dout, const Scratch& w,
                      float* sc, float* t1, float* l1, float* l2, float* m2,
                      hipStream_t s)
{
    knn_kernel<<<dim3((unsigned)(M / KNN_TS)), dim3(KNN_TS), 0, s>>>(pos, w.nbr, n);
    run_smlp(x, bp.shortcut, sc, M, di, dout, false, w, s);
    run_smlp(x, bp.mlp1, t1, M, di, dout / 8, true, w, s);
    run_lfa(bp.lfa1, w.nbr, t1, pos, l1, M, dout / 4, w, s);
    run_lfa(bp.lfa2, w.nbr, l1, pos, l2, M, dout / 2, w, s);
    run_smlp(l2, bp.mlp2, m2, M, dout / 2, dout, false, w, s);
    size_t tot = (size_t)M * dout;
    resid_leaky_kernel<<<dim3(cdiv(tot, 256)), dim3(256), 0, s>>>(m2, sc, m2, tot);
    const int nd = n / 4, Mout = M / 4;
    decim_kernel<<<dim3(cdiv((size_t)Mout * dout, 256)), dim3(256), 0, s>>>(
        m2, x_dec, n, nd, dout, Mout);
    decim_kernel<<<dim3(cdiv((size_t)Mout * 3, 256)), dim3(256), 0, s>>>(
        pos, pos_dec, n, nd, 3, Mout);
}

extern "C" void kernel_launch(void* const* d_in, const int* in_sizes, int n_in,
                              void* d_out, int out_size, void* d_ws, size_t ws_size,
                              hipStream_t stream)
{
    (void)in_sizes; (void)n_in; (void)out_size; (void)ws_size;

    const int B  = 16, N = 2048;
    const int M1 = B * N;          // 32768
    const int M2 = M1 / 4;         // 8192
    const int M3 = M2 / 4;         // 2048
    const int E1 = M1 * KNN_K;     // 524288

    // ---- parse inputs (dict order, recursive flatten) ----
    const float* x   = (const float*)d_in[0];
    const float* pos = (const float*)d_in[1];
    int ip = 2;
    auto nextp = [&]() -> const float* { return (const float*)d_in[ip++]; };
    auto getSmlp  = [&]() { SmlpP p; p.W = nextp(); p.b = nextp(); p.g = nextp(); p.beta = nextp(); return p; };
    auto getLfa   = [&]() { LfaP p; p.enc = getSmlp(); p.attW = nextp(); p.post = getSmlp(); return p; };
    auto getBlock = [&]() { BlockP p; p.mlp1 = getSmlp(); p.shortcut = getSmlp();
                            p.lfa1 = getLfa(); p.lfa2 = getLfa(); p.mlp2 = getSmlp(); return p; };

    const float* fc0W = nextp();
    const float* fc0b = nextp();
    BlockP b1 = getBlock();
    BlockP b2 = getBlock();
    SmlpP mlp1p = getSmlp();
    SmlpP end1p = getSmlp();
    const float* end2W = nextp();
    const float* end2b = nextp();

    // ---- carve workspace ----
    char* wsb = (char*)d_ws;
    size_t off = 0;
    auto allocf = [&](size_t n) -> float* { float* p = (float*)(wsb + off); off += n * sizeof(float); return p; };
    auto alloci = [&](size_t n) -> int*   { int*   p = (int*)  (wsb + off); off += n * sizeof(int);   return p; };

    float* h0     = allocf((size_t)M1 * 8);
    int*   nbr    = alloci((size_t)E1);
    float* sc     = allocf(1048576);          // max(M1*32, M2*128)
    float* t1     = allocf(131072);           // max(M1*4, M2*16)
    float* l1     = allocf(262144);           // max(M1*8, M2*32)
    float* l2     = allocf(524288);           // max(M1*16, M2*64)
    float* m2     = allocf(1048576);          // max(M1*32, M2*128)
    float* relb   = allocf((size_t)E1 * 10);  // 5.24M
    float* locse  = allocf(4194304);          // max(E1*8, E2*32)
    float* localb = allocf(8388608);          // max(E1*16, E2*64)
    float* attb   = allocf(8388608);
    float* pooled = allocf(524288);           // max(M1*16, M2*64)
    float* xa     = allocf(262144);           // b1 decimated: M2*32
    float* xb     = allocf(262144);           // b2 decimated: M3*128
    float* pos1   = allocf((size_t)M2 * 3);
    float* pos2   = allocf((size_t)M3 * 3);
    float* meanb  = allocf(128);
    float* rsigb  = allocf(128);
    float* psum   = allocf(128 * BN_SLICES);
    float* psum2  = allocf(128 * BN_SLICES);
    float* hhead  = allocf((size_t)M3 * 128);
    float* poolB  = allocf((size_t)B * 128);
    float* e1buf  = allocf((size_t)B * 32);
    float* logits = allocf((size_t)B * 40);

    Scratch w;
    w.nbr = nbr; w.rel = relb; w.locse = locse; w.local = localb;
    w.att = attb; w.pooled = pooled; w.mean = meanb; w.rsig = rsigb;
    w.psum = psum; w.psum2 = psum2;

    // ---- forward pass ----
    // fc0: [M1,3] -> [M1,8] (linear only)
    run_linear(x, fc0W, fc0b, h0, M1, 3, 8, stream);

    // block 1: di=8, do=32, n=2048 -> xa [M2,32], pos1
    run_block(b1, h0, pos, xa, pos1, M1, 2048, 8, 32, w, sc, t1, l1, l2, m2, stream);

    // block 2: di=32, do=128, n=512 -> xb [M3,128], pos2
    run_block(b2, xa, pos1, xb, pos2, M2, 512, 32, 128, w, sc, t1, l1, l2, m2, stream);

    // head: smlp 128->128, per-cloud max pool, smlp 128->32, fc 32->40, log_softmax
    run_smlp(xb, mlp1p, hhead, M3, 128, 128, true, w, stream);
    maxpool_kernel<<<dim3(cdiv((size_t)B * 128, 256)), dim3(256), 0, stream>>>(
        hhead, poolB, M3 / B, B, 128);
    run_smlp(poolB, end1p, e1buf, B, 128, 32, true, w, stream);
    run_linear(e1buf, end2W, end2b, logits, B, 32, 40, stream);
    logsoftmax_kernel<<<dim3(1), dim3(32), 0, stream>>>(logits, (float*)d_out, B, 40);
}
